// GroupedQueryAttention_3530463117421
// MI455X (gfx1250) — matrix operand order repaired and checked by reading, not compiled, not run on hardware
//
#include <hip/hip_runtime.h>

// ---------------------------------------------------------------------------
// CDNA5 (gfx1250) GQA forward: bf16 WMMA GEMMs + flash attention.
// Wave32, v_wmma_f32_16x16x32_bf16; LDS staged via async global->LDS DMA
// (ASYNCcnt) when available, pure b128 everywhere.
// ---------------------------------------------------------------------------

typedef __attribute__((ext_vector_type(16))) __bf16        v16bf;
typedef __attribute__((ext_vector_type(8)))  float         v8f;
typedef __attribute__((ext_vector_type(4)))  unsigned int  u32x4;
typedef __attribute__((ext_vector_type(4)))  int           i32x4;

union Frag {
  u32x4 u[2];
  v16bf v;
};
union Pack8 {
  u32x4 v;
  unsigned short s[8];
};

#if __has_builtin(__builtin_amdgcn_global_load_async_to_lds_b128)
#define ASYNC_COPY 1
#else
#define ASYNC_COPY 0
#endif

// 16B global -> LDS copy; async DMA (ASYNCcnt) when the builtin exists.
// Builtin signature (per hipcc diagnostic): (int4* src, int4* dst, Iint, Iint).
__device__ __forceinline__ void copy_b128(unsigned short* l, const unsigned short* g) {
#if ASYNC_COPY
  __builtin_amdgcn_global_load_async_to_lds_b128((i32x4*)g, (i32x4*)l, 0, 0);
#else
  *reinterpret_cast<u32x4*>(l) = *reinterpret_cast<const u32x4*>(g);
#endif
}

// Wait for this wave's outstanding async copies (no-op on sync fallback).
__device__ __forceinline__ void copies_commit() {
#if ASYNC_COPY
#if __has_builtin(__builtin_amdgcn_s_wait_asynccnt)
  __builtin_amdgcn_s_wait_asynccnt(0);
#else
  asm volatile("s_wait_asynccnt 0" ::: "memory");
#endif
#endif
}

__device__ __forceinline__ unsigned short f2bf(float f) {
  unsigned int u = __float_as_uint(f);
  u += 0x7FFFu + ((u >> 16) & 1u);   // round-to-nearest-even
  return (unsigned short)(u >> 16);
}

__device__ __forceinline__ v8f wmma_bf16(const Frag& a, const Frag& b, v8f c) {
  // (neg_a, A, neg_b, B, c_mod, C, reuse_a, reuse_b)
  return __builtin_amdgcn_wmma_f32_16x16x32_bf16(false, a.v, false, b.v,
                                                 (short)0, c, false, false);
}

// ---------------------------------------------------------------------------
// fp32 -> bf16 conversion (row-major copy, for X)
// ---------------------------------------------------------------------------
__global__ __launch_bounds__(256) void cvt_f32_bf16(const float* __restrict__ src,
                                                    unsigned short* __restrict__ dst,
                                                    int n) {
  int i = blockIdx.x * 256 + threadIdx.x;
  if (i < n) dst[i] = f2bf(src[i]);
}

// ---------------------------------------------------------------------------
// fp32 [K][N] -> bf16 [N][K] transpose-convert (LDS tiled, coalesced both ways)
// Grid: (N/32, K/32), block 256.
// ---------------------------------------------------------------------------
__global__ __launch_bounds__(256)
void cvt_transpose(const float* __restrict__ src, unsigned short* __restrict__ dst,
                   int K, int N) {
  __shared__ float tile[32][33];
  const int tx = threadIdx.x & 31;
  const int ty = threadIdx.x >> 5;           // 0..7
  const int n0 = blockIdx.x * 32;
  const int k0 = blockIdx.y * 32;
#pragma unroll
  for (int i = 0; i < 4; ++i)
    tile[ty + 8 * i][tx] = src[(size_t)(k0 + ty + 8 * i) * N + n0 + tx];
  __syncthreads();
#pragma unroll
  for (int i = 0; i < 4; ++i)
    dst[(size_t)(n0 + ty + 8 * i) * K + k0 + tx] = f2bf(tile[tx][ty + 8 * i]);
}

// ---------------------------------------------------------------------------
// Tiled bf16 WMMA GEMM: C[M,N] = A[M,K] @ Bt[N,K]^T  (Bt is pre-transposed).
// Block = 256 threads (8 wave32), block tile 64x256, BK = 32.  Double-buffered
// LDS fed by async global->LDS DMA: stage chunk t+1 while WMMAs run on chunk t.
// Each wave: 32x64 output = 2x4 WMMA tiles -> 8 WMMAs per K-chunk.
// OUTMODE: 0 = bf16 row-major, 1 = bf16 transposed [N][M], 2 = f32 + bias.
// ---------------------------------------------------------------------------
template <int OUTMODE>
__global__ __launch_bounds__(256)
void gemm_wmma(const unsigned short* __restrict__ A,    // [M][K] bf16
               const unsigned short* __restrict__ Bt,   // [N][K] bf16
               unsigned short* __restrict__ Cb,
               float* __restrict__ Cf,
               const float* __restrict__ bias,
               int M, int N, int K) {
  __shared__ alignas(16) unsigned short As[2][64][32];
  __shared__ alignas(16) unsigned short Bs[2][256][40];  // 80B rows: 16B aligned

  const int tid  = threadIdx.x;
  const int lane = tid & 31;
  const int wave = tid >> 5;
  const int wm   = (wave >> 2) * 32;   // wave M offset
  const int wn   = (wave & 3) * 64;    // wave N offset
  const int m0   = blockIdx.y * 64;
  const int n0   = blockIdx.x * 256;
  const int col  = lane & 15;
  const int kh   = lane >> 4;

  v8f acc[2][4];
#pragma unroll
  for (int i = 0; i < 2; ++i)
#pragma unroll
    for (int j = 0; j < 4; ++j)
#pragma unroll
      for (int e = 0; e < 8; ++e) acc[i][j][e] = 0.0f;

  const int ar = tid >> 2;          // 0..63
  const int ac = (tid & 3) * 8;     // 0,8,16,24
  const unsigned short* Ag = A + (size_t)(m0 + ar) * K + ac;
  const unsigned short* Bg = Bt + (size_t)(n0 + tid) * K;

  const int T = K >> 5;

  // Stage one 64x32 A tile + 256x32 B tile into LDS buffer `bu`.
  auto stage = [&](int bu, int k0) {
    copy_b128(&As[bu][ar][ac], Ag + k0);
#pragma unroll
    for (int j = 0; j < 4; ++j)
      copy_b128(&Bs[bu][tid][j * 8], Bg + k0 + j * 8);
  };

  stage(0, 0);
  copies_commit();
  __syncthreads();

  for (int t = 0; t < T; ++t) {
    const int cur = t & 1;
    if (t + 1 < T) stage(cur ^ 1, (t + 1) * 32);   // async DMA overlaps WMMAs

    // A fragments: ISA 16-bit A 16x32 layout (K interleaved by lane half).
    Frag a[2], b[4];
#pragma unroll
    for (int mi = 0; mi < 2; ++mi) {
      const int r = wm + mi * 16 + col;
      a[mi].u[0] = *reinterpret_cast<const u32x4*>(&As[cur][r][kh * 8]);
      a[mi].u[1] = *reinterpret_cast<const u32x4*>(&As[cur][r][16 + kh * 8]);
    }
    // B fragments: same K interleave as A (lane half kh: K 8kh..8kh+7, 16+8kh..16+8kh+7).
#pragma unroll
    for (int ni = 0; ni < 4; ++ni) {
      const int c = wn + ni * 16 + col;
      b[ni].u[0] = *reinterpret_cast<const u32x4*>(&Bs[cur][c][kh * 8]);
      b[ni].u[1] = *reinterpret_cast<const u32x4*>(&Bs[cur][c][16 + kh * 8]);
    }
#pragma unroll
    for (int mi = 0; mi < 2; ++mi)
#pragma unroll
      for (int ni = 0; ni < 4; ++ni)
        acc[mi][ni] = wmma_bf16(a[mi], b[ni], acc[mi][ni]);

    copies_commit();     // this wave's async copies landed
    __syncthreads();     // everyone's landed; cur buffer free for reuse
  }

  // Epilogue. C layout: VGPR r -> row r + 8*kh, lane col -> column.
#pragma unroll
  for (int mi = 0; mi < 2; ++mi)
#pragma unroll
    for (int ni = 0; ni < 4; ++ni) {
      const int gn = n0 + wn + ni * 16 + col;
      if (OUTMODE == 1) {
        // Transposed bf16 store: 8 consecutive rows -> one b128 per lane.
        Pack8 p;
#pragma unroll
        for (int r = 0; r < 8; ++r) p.s[r] = f2bf(acc[mi][ni][r]);
        const int gmBase = m0 + wm + mi * 16 + 8 * kh;
        *reinterpret_cast<u32x4*>(Cb + (size_t)gn * M + gmBase) = p.v;
      } else {
#pragma unroll
        for (int r = 0; r < 8; ++r) {
          const int gm = m0 + wm + mi * 16 + r + 8 * kh;
          const float v = acc[mi][ni][r];
          if (OUTMODE == 2) Cf[(size_t)gm * N + gn] = v + bias[gn];
          else              Cb[(size_t)gm * N + gn] = f2bf(v);
        }
      }
    }
}

// ---------------------------------------------------------------------------
// Flash-attention with causal mask, GQA (4 query heads share one K/V head).
// Grid: (32 query tiles of 64, 16 heads, 2 batches). Block: 128 threads.
// Each of 4 waves owns 16 query rows; key tiles of 32.  V arrives already
// transposed ([dkv][b*n]); K/V tiles staged by async global->LDS DMA.
// ---------------------------------------------------------------------------
#define SEQ    2048
#define DMODEL 2048
#define DKV    512
#define BN_TOT 4096
#define HDIM   128

__global__ __launch_bounds__(128)
void gqa_flash(const unsigned short* __restrict__ Q,    // [B*SEQ][DMODEL] bf16
               const unsigned short* __restrict__ Kp,   // [B*SEQ][DKV]    bf16
               const unsigned short* __restrict__ Vt,   // [DKV][B*SEQ]    bf16
               unsigned short* __restrict__ Ctx) {      // [B*SEQ][DMODEL] bf16
  __shared__ alignas(16) unsigned short Kt[32][128];    // key tile, row-major
  __shared__ alignas(16) unsigned short Vl[128][40];    // value tile [dim][key]
  __shared__ alignas(16) unsigned short Pb[4][16][32];  // per-wave P bounce

  const int qt    = blockIdx.x;
  const int head  = blockIdx.y;
  const int batch = blockIdx.z;
  const int grp   = head >> 2;            // KV group (GROUP_SIZE = 4)
  const int tid   = threadIdx.x;
  const int lane  = tid & 31;
  const int wave  = tid >> 5;
  const int col   = lane & 15;
  const int kh    = lane >> 4;
  const int q0    = qt * 64 + wave * 16;  // first query row for this wave
  const size_t rowbase = (size_t)batch * SEQ;

  // Q fragments for this wave's 16 rows: 16x128 = 4 A-frags in VGPRs.
  Frag qf[4];
  {
    const unsigned short* qr = Q + (rowbase + q0 + col) * DMODEL + head * HDIM;
#pragma unroll
    for (int c = 0; c < 4; ++c) {
      qf[c].u[0] = *reinterpret_cast<const u32x4*>(qr + c * 32 + kh * 8);
      qf[c].u[1] = *reinterpret_cast<const u32x4*>(qr + c * 32 + 16 + kh * 8);
    }
  }

  v8f ctx[8];
#pragma unroll
  for (int nt = 0; nt < 8; ++nt)
#pragma unroll
    for (int e = 0; e < 8; ++e) ctx[nt][e] = 0.0f;

  float m_run[8], l_run[8];
#pragma unroll
  for (int r = 0; r < 8; ++r) { m_run[r] = -1e30f; l_run[r] = 0.0f; }

  const float scale = 0.08838834764831845f;   // 1/sqrt(128)
  const float L2E   = 1.4426950408889634f;
  const int ntiles  = qt * 2 + 2;             // causal: keys up to qt*64+63

  // Per-thread staging addresses.
  const int kr = tid >> 2;                    // 0..31  (K tile row)
  const int kc = (tid & 3) * 32;              // 0,32,64,96 (head-dim chunk)
  const unsigned short* Kg = Kp + (rowbase + kr) * DKV + grp * HDIM + kc;
  const unsigned short* Vg = Vt + (size_t)(grp * HDIM + tid) * BN_TOT + rowbase;

  for (int t = 0; t < ntiles; ++t) {
    const int j0 = t * 32;
    // Stage K tile (32x128, row-major) and V tile (128 dims x 32 keys)
    // via async global->LDS DMA (no VGPR round-trip).
#pragma unroll
    for (int j = 0; j < 4; ++j)
      copy_b128(&Kt[kr][kc + j * 8], Kg + (size_t)j0 * DKV + j * 8);
#pragma unroll
    for (int j = 0; j < 4; ++j)
      copy_b128(&Vl[tid][j * 8], Vg + j0 + j * 8);
    if (t + 1 < ntiles) {           // prefetch next tiles
      __builtin_prefetch(Kg + (size_t)(j0 + 32) * DKV, 0, 3);
      __builtin_prefetch(Vg + j0 + 32, 0, 3);
    }
    copies_commit();
    __syncthreads();

    if (j0 <= q0 + 15) {                      // wave-uniform causal skip
      // Scores S[16x32] = Q(16x128) @ K^T(128x32).
      v8f s0, s1;
#pragma unroll
      for (int e = 0; e < 8; ++e) { s0[e] = 0.0f; s1[e] = 0.0f; }
#pragma unroll
      for (int c = 0; c < 4; ++c) {
        Frag kf0, kf1;   // B column j = key row j -> contiguous over head-dim
        kf0.u[0] = *reinterpret_cast<const u32x4*>(&Kt[col][c * 32 + kh * 8]);
        kf0.u[1] = *reinterpret_cast<const u32x4*>(&Kt[col][c * 32 + 16 + kh * 8]);
        kf1.u[0] = *reinterpret_cast<const u32x4*>(&Kt[16 + col][c * 32 + kh * 8]);
        kf1.u[1] = *reinterpret_cast<const u32x4*>(&Kt[16 + col][c * 32 + 16 + kh * 8]);
        s0 = wmma_bf16(qf[c], kf0, s0);
        s1 = wmma_bf16(qf[c], kf1, s1);
      }

      // Online softmax. Row r+8*kh lives in one 16-lane half; xor masks
      // 1/2/4/8 reduce within the half without crossing rows.
      float alpha[8];
#pragma unroll
      for (int r = 0; r < 8; ++r) {
        const int qrow = q0 + r + 8 * kh;
        float v0 = (j0 + col      <= qrow) ? s0[r] * scale : -1e30f;
        float v1 = (j0 + 16 + col <= qrow) ? s1[r] * scale : -1e30f;
        float mx = fmaxf(v0, v1);
#pragma unroll
        for (int d = 1; d < 16; d <<= 1) mx = fmaxf(mx, __shfl_xor(mx, d));
        const float mnew = fmaxf(m_run[r], mx);
        const float a    = exp2f((m_run[r] - mnew) * L2E);
        const float p0   = exp2f((v0 - mnew) * L2E);
        const float p1   = exp2f((v1 - mnew) * L2E);
        float ps = p0 + p1;
#pragma unroll
        for (int d = 1; d < 16; d <<= 1) ps += __shfl_xor(ps, d);
        l_run[r] = l_run[r] * a + ps;
        m_run[r] = mnew;
        alpha[r] = a;
        // C-layout -> LDS so we can refetch as an A fragment.
        Pb[wave][r + 8 * kh][col]      = f2bf(p0);
        Pb[wave][r + 8 * kh][16 + col] = f2bf(p1);
      }
#pragma unroll
      for (int nt = 0; nt < 8; ++nt)
#pragma unroll
        for (int r = 0; r < 8; ++r) ctx[nt][r] *= alpha[r];

      // DS ops are in-order per wave; wait + compiler barrier before refetch.
      asm volatile("s_wait_dscnt 0" ::: "memory");

      Frag pf;   // P as 16x32 A fragment
      pf.u[0] = *reinterpret_cast<const u32x4*>(&Pb[wave][col][kh * 8]);
      pf.u[1] = *reinterpret_cast<const u32x4*>(&Pb[wave][col][16 + kh * 8]);
#pragma unroll
      for (int nt = 0; nt < 8; ++nt) {
        Frag vf; // B column = output dim -> contiguous over keys in Vl
        vf.u[0] = *reinterpret_cast<const u32x4*>(&Vl[nt * 16 + col][kh * 8]);
        vf.u[1] = *reinterpret_cast<const u32x4*>(&Vl[nt * 16 + col][16 + kh * 8]);
        ctx[nt] = wmma_bf16(pf, vf, ctx[nt]);
      }
    }
    __syncthreads();
  }

  // Normalize and store ctx (bf16) at this head's slice.
#pragma unroll
  for (int r = 0; r < 8; ++r) {
    const float inv = 1.0f / l_run[r];
    const size_t orow = (rowbase + q0 + r + 8 * kh) * DMODEL + head * HDIM;
#pragma unroll
    for (int nt = 0; nt < 8; ++nt)
      Ctx[orow + nt * 16 + col] = f2bf(ctx[nt][r] * inv);
  }
}

// ---------------------------------------------------------------------------
// Launcher
// ---------------------------------------------------------------------------
extern "C" void kernel_launch(void* const* d_in, const int* in_sizes, int n_in,
                              void* d_out, int out_size, void* d_ws, size_t ws_size,
                              hipStream_t stream) {
  const float* x  = (const float*)d_in[0];
  const float* Wq = (const float*)d_in[1];
  const float* Wk = (const float*)d_in[2];
  const float* Wv = (const float*)d_in[3];
  const float* Wo = (const float*)d_in[4];
  const float* bo = (const float*)d_in[5];
  float* out = (float*)d_out;

  const size_t nX  = 4096ull * 2048ull;   // b*n x D_IN
  const size_t nWq = 2048ull * 2048ull;
  const size_t nWk = 2048ull * 512ull;
  const size_t nKp = 4096ull * 512ull;

  unsigned short* ws  = (unsigned short*)d_ws;
  unsigned short* Xb  = ws;               // [4096][2048]
  unsigned short* Wqt = Xb  + nX;         // [2048 out][2048 in]  (transposed)
  unsigned short* Wkt = Wqt + nWq;        // [512][2048]
  unsigned short* Wvt = Wkt + nWk;        // [512][2048]
  unsigned short* Wot = Wvt + nWk;        // [2048][2048]
  unsigned short* Qb  = Wot + nWq;        // [4096][2048]
  unsigned short* Kb  = Qb  + nX;         // [4096][512]
  unsigned short* Vtb = Kb  + nKp;        // [512][4096]  (transposed)
  unsigned short* Cxb = Vtb + nKp;        // [4096][2048]

  // Staging: X row-major; weights transpose-converted to [out][in].
  cvt_f32_bf16<<<dim3((unsigned)((nX + 255) / 256)), dim3(256), 0, stream>>>(
      x, Xb, (int)nX);
  cvt_transpose<<<dim3(64, 64), dim3(256), 0, stream>>>(Wq, Wqt, 2048, 2048);
  cvt_transpose<<<dim3(16, 64), dim3(256), 0, stream>>>(Wk, Wkt, 2048, 512);
  cvt_transpose<<<dim3(16, 64), dim3(256), 0, stream>>>(Wv, Wvt, 2048, 512);
  cvt_transpose<<<dim3(64, 64), dim3(256), 0, stream>>>(Wo, Wot, 2048, 2048);

  // Projections (bf16 WMMA). V written transposed for the attention kernel.
  gemm_wmma<0><<<dim3(8, 64), dim3(256), 0, stream>>>(
      Xb, Wqt, Qb, nullptr, nullptr, 4096, 2048, 2048);
  gemm_wmma<0><<<dim3(2, 64), dim3(256), 0, stream>>>(
      Xb, Wkt, Kb, nullptr, nullptr, 4096, 512, 2048);
  gemm_wmma<1><<<dim3(2, 64), dim3(256), 0, stream>>>(
      Xb, Wvt, Vtb, nullptr, nullptr, 4096, 512, 2048);

  // Causal GQA flash attention.
  gqa_flash<<<dim3(32, 16, 2), dim3(128), 0, stream>>>(Qb, Kb, Vtb, Cxb);

  // Output projection + bias, fp32 out.
  gemm_wmma<2><<<dim3(8, 64), dim3(256), 0, stream>>>(
      Cxb, Wot, nullptr, out, bo, 4096, 2048, 2048);
}
